// Attention_50946902065463
// MI455X (gfx1250) — matrix operand order repaired and checked by reading, not compiled, not run on hardware
//
#include <hip/hip_runtime.h>
#include <hip/hip_bf16.h>

// ---------------------------------------------------------------------------
// Problem constants (from reference)
// ---------------------------------------------------------------------------
#define N_EMBD   2048
#define N_HEAD   16
#define HEAD_DIM 128
#define BATCH    2
#define SEQ      2048
#define MTOT     (BATCH * SEQ)        // 4096 rows
#define N_QKV    (3 * N_EMBD)         // 6144
#define SM_SCALE 0.08838834764831845f // 1/sqrt(HEAD_DIM)

typedef __attribute__((ext_vector_type(16))) __bf16 v16bf;
typedef __attribute__((ext_vector_type(8)))  __bf16 v8bf;
typedef __attribute__((ext_vector_type(8)))  float  v8f;
typedef __attribute__((ext_vector_type(4)))  unsigned int u32x4;
typedef __attribute__((ext_vector_type(8)))  int i32x8;
typedef __attribute__((ext_vector_type(4)))  int i32x4;

// ---------------------------------------------------------------------------
// Fragment helpers — layouts per CDNA5 ISA 7.12.2 (wave32)
//
// A (16x32 bf16): lane r=lane%16 -> row M=r, half h=lane/16.
//   elements 0..7  hold K = 8h + (0..7)      (contiguous)
//   elements 8..15 hold K = 16 + 8h + (0..7) (contiguous)
// B (32x16 bf16): lane r -> column N=r, half h.
//   elements 0..7 hold K = 8h + (0..7), elements 8..15 hold K = 16 + 8h + (0..7)
// C/D (16x16 f32): vgpr g -> row M = g + 8h, column N = r.
// ---------------------------------------------------------------------------
__device__ __forceinline__ v16bf make_frag(v8bf lo, v8bf hi) {
  v16bf f;
#pragma unroll
  for (int i = 0; i < 8; ++i) { f[i] = lo[i]; f[i + 8] = hi[i]; }
  return f;
}

__device__ __forceinline__ v16bf load_a_frag(const __bf16* row, int h) {
  v8bf lo = *(const v8bf*)(row + 8 * h);
  v8bf hi = *(const v8bf*)(row + 16 + 8 * h);
  return make_frag(lo, hi);
}

__device__ __forceinline__ v16bf load_b_frag(const __bf16* col, int h) {
  v8bf lo = *(const v8bf*)(col + 8 * h);
  v8bf hi = *(const v8bf*)(col + 16 + 8 * h);
  return make_frag(lo, hi);
}

__device__ __forceinline__ v8f wmma_bf16(v16bf a, v16bf b, v8f c) {
  return __builtin_amdgcn_wmma_f32_16x16x32_bf16(
      /*neg_a=*/false, a, /*neg_b=*/false, b,
      /*c_mod=*/(short)0, c, /*reuse_a=*/false, /*reuse_b=*/false);
}

// ---------------------------------------------------------------------------
// Pre-pass kernels
// ---------------------------------------------------------------------------
__global__ void cvt_f32_bf16(const float* __restrict__ in,
                             __bf16* __restrict__ out, int n) {
  int i = blockIdx.x * blockDim.x + threadIdx.x;
  int stride = gridDim.x * blockDim.x;
  for (; i < n; i += stride) out[i] = (__bf16)in[i];
}

// in[k*N + n] (row-major KxN, fp32) -> out[n*K + k] (N-major, bf16)
__global__ __launch_bounds__(256) void transpose_cvt(
    const float* __restrict__ in, __bf16* __restrict__ out, int K, int N) {
  __shared__ __bf16 tile[32][33];
  const int tx = threadIdx.x & 31, ty = threadIdx.x >> 5;
  const int nb = N >> 5;
  const int k0 = (blockIdx.x / nb) << 5;
  const int n0 = (blockIdx.x % nb) << 5;
#pragma unroll
  for (int i = 0; i < 32; i += 8)
    tile[ty + i][tx] = (__bf16)in[(size_t)(k0 + ty + i) * N + n0 + tx];
  __syncthreads();
#pragma unroll
  for (int i = 0; i < 32; i += 8)
    out[(size_t)(n0 + ty + i) * K + k0 + tx] = tile[tx][ty + i];
}

// ---------------------------------------------------------------------------
// Tensor Data Mover 2D tile load (6-arg builtin on this toolchain).
//   rowlen : tensor row length == dim0 stride (elements)
//   tile0  : tile width  (elements, contiguous in memory)
//   tile1  : tile height (rows)
//   padI   : D#.pad_interval code (3 -> pad every 64B, 5 -> every 256B);
//            pad_amount fixed at 16B -> LDS row stride = rowbytes + 16.
// ---------------------------------------------------------------------------
#define HAVE_TDM __has_builtin(__builtin_amdgcn_tensor_load_to_lds)

#if HAVE_TDM
__device__ __forceinline__ void tdm_load_2d(const __bf16* gptr, void* lptr,
                                            unsigned rowlen, unsigned tile0,
                                            unsigned tile1, unsigned padI) {
  const unsigned long long ga = (unsigned long long)(size_t)gptr;
  u32x4 g0;
  g0[0] = 1u;                                        // count=1, user mode
  g0[1] = (unsigned)(size_t)lptr;                    // LDS byte address
  g0[2] = (unsigned)(ga & 0xFFFFFFFFu);              // global_addr[31:0]
  g0[3] = (unsigned)((ga >> 32) & 0x1FFFFFFu)        // global_addr[56:32]
        | (2u << 30);                                // type = 2 (image)
  i32x8 g1;
  g1[0] = (int)((1u << 16)                           // data_size = 2 bytes
        | (1u << 20)                                 // pad_enable
        | (padI << 22)                               // pad_interval
        | (3u << 25));                               // pad_amount: 4 DW (16B)
  g1[1] = (int)((rowlen & 0xFFFFu) << 16);           // tensor_dim0 lo16
  g1[2] = (int)((rowlen >> 16) | (tile1 << 16));     // dim0 hi | tensor_dim1 lo
  g1[3] = (int)((tile1 >> 16) | (tile0 << 16));      // dim1 hi | tile_dim0
  g1[4] = (int)tile1;                                // tile_dim1, tile_dim2=0
  g1[5] = (int)rowlen;                               // tensor_dim0_stride lo
  g1[6] = 0;
  g1[7] = 0;
  const i32x4 z4 = {0, 0, 0, 0};
  const i32x8 z8 = {0, 0, 0, 0, 0, 0, 0, 0};
  __builtin_amdgcn_tensor_load_to_lds(g0, g1, z4, z4, z8, 0);
}
#endif

// ---------------------------------------------------------------------------
// LDS-tiled 128x128 GEMM core (K = 2048, BK = 32), triple-buffered TDM fill,
// ONE barrier per K-step: wait tensorcnt<=2 (tile t landed) -> barrier ->
// issue tile t+2 (its buffer's last readers drained at the barrier) -> compute.
// ---------------------------------------------------------------------------
#define BK      32
#define LDS_PAD 8     // row stride 40 elems (80 B, 16B-aligned)

__device__ __forceinline__ void gemm_core_128x128(
    const __bf16* __restrict__ Ag,   // base of 128 block rows, stride K
    const __bf16* __restrict__ Bg,   // base of 128 block cols (N-major), stride K
    int K, v8f acc[2][4]) {
  const int tid  = threadIdx.x;
  const int lane = tid & 31;
  const int r = lane & 15, h = lane >> 4;
  const int wave = tid >> 5;
  const int wm = (wave & 3) * 32;    // wave's M offset inside block tile
  const int wn = (wave >> 2) * 64;   // wave's N offset inside block tile
  const int nk = K / BK;

#if HAVE_TDM
  __shared__ __bf16 lA[3][128][BK + LDS_PAD];
  __shared__ __bf16 lB[3][128][BK + LDS_PAD];
  const bool issuer = (wave == 0);
  if (issuer) {
    tdm_load_2d(Ag,      &lA[0][0][0], 2048, 32, 128, 3);
    tdm_load_2d(Bg,      &lB[0][0][0], 2048, 32, 128, 3);
    tdm_load_2d(Ag + BK, &lA[1][0][0], 2048, 32, 128, 3);
    tdm_load_2d(Bg + BK, &lB[1][0][0], 2048, 32, 128, 3);
  }
  int buf = 0, nb = 2;
  for (int t = 0; t < nk; ++t) {
    if (issuer) {
      if (t + 1 < nk) __builtin_amdgcn_s_wait_tensorcnt(2);
      else            __builtin_amdgcn_s_wait_tensorcnt(0);
    }
    __syncthreads();                            // publish tile t; drain readers
    if (issuer && t + 2 < nk) {
      tdm_load_2d(Ag + (t + 2) * BK, &lA[nb][0][0], 2048, 32, 128, 3);
      tdm_load_2d(Bg + (t + 2) * BK, &lB[nb][0][0], 2048, 32, 128, 3);
    }

    v16bf am0 = load_a_frag(&lA[buf][wm + r][0], h);
    v16bf am1 = load_a_frag(&lA[buf][wm + 16 + r][0], h);
    v16bf bn[4];
#pragma unroll
    for (int nt = 0; nt < 4; ++nt)
      bn[nt] = load_b_frag(&lB[buf][wn + nt * 16 + r][0], h);
#pragma unroll
    for (int nt = 0; nt < 4; ++nt) {
      acc[0][nt] = wmma_bf16(am0, bn[nt], acc[0][nt]);
      acc[1][nt] = wmma_bf16(am1, bn[nt], acc[1][nt]);
    }
    buf = (buf == 2) ? 0 : buf + 1;
    nb  = (nb  == 2) ? 0 : nb  + 1;
  }
#else
  // ---- fallback: per-thread global->reg->LDS staging, double buffered ----
  __shared__ __bf16 lA[2][128][BK + LDS_PAD];
  __shared__ __bf16 lB[2][128][BK + LDS_PAD];
  const int row0 = tid >> 2;
  const int row1 = row0 + 64;
  const int seg  = (tid & 3) * 8;

  v8bf ra0, ra1, rb0, rb1;
  auto load_gl = [&](int k0) {
    ra0 = *(const v8bf*)(Ag + (size_t)row0 * K + k0 + seg);
    ra1 = *(const v8bf*)(Ag + (size_t)row1 * K + k0 + seg);
    rb0 = *(const v8bf*)(Bg + (size_t)row0 * K + k0 + seg);
    rb1 = *(const v8bf*)(Bg + (size_t)row1 * K + k0 + seg);
  };
  auto store_lds = [&](int b) {
    *(v8bf*)(&lA[b][row0][seg]) = ra0;
    *(v8bf*)(&lA[b][row1][seg]) = ra1;
    *(v8bf*)(&lB[b][row0][seg]) = rb0;
    *(v8bf*)(&lB[b][row1][seg]) = rb1;
  };

  load_gl(0);
  store_lds(0);
  __syncthreads();

  for (int t = 0; t < nk; ++t) {
    const int buf = t & 1;
    if (t + 1 < nk) load_gl((t + 1) * BK);

    v16bf am0 = load_a_frag(&lA[buf][wm + r][0], h);
    v16bf am1 = load_a_frag(&lA[buf][wm + 16 + r][0], h);
    v16bf bn[4];
#pragma unroll
    for (int nt = 0; nt < 4; ++nt)
      bn[nt] = load_b_frag(&lB[buf][wn + nt * 16 + r][0], h);
#pragma unroll
    for (int nt = 0; nt < 4; ++nt) {
      acc[0][nt] = wmma_bf16(am0, bn[nt], acc[0][nt]);
      acc[1][nt] = wmma_bf16(am1, bn[nt], acc[1][nt]);
    }

    if (t + 1 < nk) store_lds(buf ^ 1);
    __syncthreads();
  }
#endif
}

// ---------------------------------------------------------------------------
// QKV projection: [4096,2048] @ [2048,6144] + bias.
// Scatters Q (pre-scaled by 1/sqrt(D)) -> [B,H,S,D], K -> [B,H,S,D],
// V -> transposed [B,H,D,S].
// ---------------------------------------------------------------------------
__global__ __launch_bounds__(256) void qkv_gemm(
    const __bf16* __restrict__ X, const __bf16* __restrict__ Wt,
    const float* __restrict__ bias, __bf16* __restrict__ qws,
    __bf16* __restrict__ kws, __bf16* __restrict__ vtws) {
  const int nbN = N_QKV / 128;
  const int m0 = (blockIdx.x / nbN) * 128;
  const int n0 = (blockIdx.x % nbN) * 128;

  v8f acc[2][4] = {};
  gemm_core_128x128(X + (size_t)m0 * N_EMBD, Wt + (size_t)n0 * N_EMBD,
                    N_EMBD, acc);

  const int lane = threadIdx.x & 31;
  const int r = lane & 15, h = lane >> 4;
  const int wave = threadIdx.x >> 5;
  const int wm = (wave & 3) * 32, wn = (wave >> 2) * 64;

#pragma unroll
  for (int mi = 0; mi < 2; ++mi) {
#pragma unroll
    for (int nt = 0; nt < 4; ++nt) {
      const int n = n0 + wn + nt * 16 + r;
      const float bv = bias[n];
#pragma unroll
      for (int g = 0; g < 8; ++g) {
        const int m = m0 + wm + mi * 16 + g + 8 * h;
        const int s = m & (SEQ - 1);
        const int bb = m >> 11;
        const float val = acc[mi][nt][g] + bv;
        if (n < N_EMBD) {
          const int head = n >> 7, d = n & 127;
          qws[(((size_t)bb * N_HEAD + head) * SEQ + s) * HEAD_DIM + d] =
              (__bf16)(val * SM_SCALE);
        } else if (n < 2 * N_EMBD) {
          const int n2 = n - N_EMBD, head = n2 >> 7, d = n2 & 127;
          kws[(((size_t)bb * N_HEAD + head) * SEQ + s) * HEAD_DIM + d] =
              (__bf16)val;
        } else {
          const int n3 = n - 2 * N_EMBD, head = n3 >> 7, d = n3 & 127;
          vtws[(((size_t)bb * N_HEAD + head) * HEAD_DIM + d) * SEQ + s] =
              (__bf16)val;
        }
      }
    }
  }
}

// ---------------------------------------------------------------------------
// Flash attention. Block = 8 waves = 128 consecutive query rows of one
// (batch, head). K/V tiles for each 32-key block are TDM-DMA'd into
// triple-buffered LDS ONCE per block (8x traffic reduction vs per-wave
// fetch); waves past their causal bound skip compute but keep barriers.
//   kT: 32 key rows x 128 d  (rows 256B, pad 16B -> 272B stride)
//   vT: 128 d rows x 32 keys (rows  64B, pad 16B ->  80B stride)
// ---------------------------------------------------------------------------
__global__ __launch_bounds__(256) void attn_kernel(
    const __bf16* __restrict__ qg, const __bf16* __restrict__ kg,
    const __bf16* __restrict__ vtg, __bf16* __restrict__ og) {
  __shared__ __bf16 plds[8][16 * 32];

  const int wave = threadIdx.x >> 5;
  const int lane = threadIdx.x & 31;
  const int r = lane & 15, h = lane >> 4;
  const int wid = blockIdx.x * 8 + wave;
  const int q0 = (wid & (SEQ / 16 - 1)) * 16;
  const int bh = wid >> 7;

  const __bf16* Q  = qg  + (size_t)bh * SEQ * HEAD_DIM;
  const __bf16* K  = kg  + (size_t)bh * SEQ * HEAD_DIM;
  const __bf16* Vt = vtg + (size_t)bh * HEAD_DIM * SEQ;

  v16bf qf[4];
#pragma unroll
  for (int j = 0; j < 4; ++j)
    qf[j] = load_a_frag(Q + (size_t)(q0 + r) * HEAD_DIM + 32 * j, h);

  v8f o[8];
#pragma unroll
  for (int nt = 0; nt < 8; ++nt) o[nt] = (v8f){};
  float mrow[8], lrow[8];
#pragma unroll
  for (int g = 0; g < 8; ++g) { mrow[g] = -__builtin_inff(); lrow[g] = 0.f; }

  const int my_end = q0 + 16;       // this wave's causal key bound

#if HAVE_TDM
  __shared__ __bf16 kT[3][32][136];
  __shared__ __bf16 vT[3][128][40];

  const int q0_blk  = (blockIdx.x & 15) * 128;   // block's first query row
  const int ntiles  = (q0_blk + 128) / 32;       // uniform across block
  const bool issuer = (wave == 0);
  if (issuer) {
    tdm_load_2d(K,                  &kT[0][0][0], 128, 128, 32, 5);
    tdm_load_2d(Vt,                 &vT[0][0][0], 2048, 32, 128, 3);
    tdm_load_2d(K + 32 * HEAD_DIM,  &kT[1][0][0], 128, 128, 32, 5);
    tdm_load_2d(Vt + 32,            &vT[1][0][0], 2048, 32, 128, 3);
  }
  int buf = 0, nxt = 2;
  for (int t = 0; t < ntiles; ++t) {
    const int kb = 32 * t;
    if (issuer) {
      if (t + 1 < ntiles) __builtin_amdgcn_s_wait_tensorcnt(2);
      else                __builtin_amdgcn_s_wait_tensorcnt(0);
    }
    __syncthreads();                 // publish tile t; drain prior readers
    if (issuer && t + 2 < ntiles) {
      tdm_load_2d(K + (size_t)(kb + 64) * HEAD_DIM, &kT[nxt][0][0],
                  128, 128, 32, 5);
      tdm_load_2d(Vt + (kb + 64), &vT[nxt][0][0], 2048, 32, 128, 3);
    }

    if (kb < my_end) {               // uniform per wave (EXEC all-1s inside)
      v16bf kf[8];
#pragma unroll
      for (int t2 = 0; t2 < 2; ++t2)
#pragma unroll
        for (int j = 0; j < 4; ++j)
          kf[t2 * 4 + j] = load_b_frag(&kT[buf][16 * t2 + r][32 * j], h);

      v8f sc0 = (v8f){}, sc1 = (v8f){};
#pragma unroll
      for (int j = 0; j < 4; ++j) sc0 = wmma_bf16(qf[j], kf[j], sc0);
#pragma unroll
      for (int j = 0; j < 4; ++j) sc1 = wmma_bf16(qf[j], kf[4 + j], sc1);

      v16bf vf[8];
#pragma unroll
      for (int nt = 0; nt < 8; ++nt)
        vf[nt] = load_b_frag(&vT[buf][16 * nt + r][0], h);

      const int key0 = kb + r, key1 = kb + 16 + r;
#pragma unroll
      for (int g = 0; g < 8; ++g) {
        const int query = q0 + g + 8 * h;
        float v0 = sc0[g], v1 = sc1[g];
        if (key0 > query) v0 = -__builtin_inff();
        if (key1 > query) v1 = -__builtin_inff();

        float mx = fmaxf(v0, v1);
        mx = fmaxf(mx, __shfl_xor(mx, 1, 32));
        mx = fmaxf(mx, __shfl_xor(mx, 2, 32));
        mx = fmaxf(mx, __shfl_xor(mx, 4, 32));
        mx = fmaxf(mx, __shfl_xor(mx, 8, 32));

        const float mnew = fmaxf(mrow[g], mx);
        const float p0 = __expf(v0 - mnew);
        const float p1 = __expf(v1 - mnew);
        float rs = p0 + p1;
        rs += __shfl_xor(rs, 1, 32);
        rs += __shfl_xor(rs, 2, 32);
        rs += __shfl_xor(rs, 4, 32);
        rs += __shfl_xor(rs, 8, 32);

        const float scl = __expf(mrow[g] - mnew);
        lrow[g] = lrow[g] * scl + rs;
        mrow[g] = mnew;
#pragma unroll
        for (int nt = 0; nt < 8; ++nt) o[nt][g] *= scl;

        const int row = g + 8 * h;
        plds[wave][row * 32 + r]      = (__bf16)p0;
        plds[wave][row * 32 + 16 + r] = (__bf16)p1;
      }

      asm volatile("s_wait_dscnt 0" ::: "memory");
      v16bf pf = load_a_frag(&plds[wave][r * 32], h);

#pragma unroll
      for (int nt = 0; nt < 8; ++nt) o[nt] = wmma_bf16(pf, vf[nt], o[nt]);
      asm volatile("" ::: "memory");
    }
    buf = (buf == 2) ? 0 : buf + 1;
    nxt = (nxt == 2) ? 0 : nxt + 1;
  }
#else
  // ---- fallback: per-wave global fetch of K/V fragments ----
  for (int kb = 0; kb < my_end; kb += 32) {
    v16bf kf[8];
#pragma unroll
    for (int t = 0; t < 2; ++t)
#pragma unroll
      for (int j = 0; j < 4; ++j)
        kf[t * 4 + j] =
            load_b_frag(K + (size_t)(kb + 16 * t + r) * HEAD_DIM + 32 * j, h);

    v8f sc0 = (v8f){}, sc1 = (v8f){};
#pragma unroll
    for (int j = 0; j < 4; ++j) sc0 = wmma_bf16(qf[j], kf[j], sc0);
#pragma unroll
    for (int j = 0; j < 4; ++j) sc1 = wmma_bf16(qf[j], kf[4 + j], sc1);

    v16bf vf[8];
#pragma unroll
    for (int nt = 0; nt < 8; ++nt)
      vf[nt] = load_b_frag(Vt + (size_t)(16 * nt + r) * SEQ + kb, h);

    const int key0 = kb + r, key1 = kb + 16 + r;
#pragma unroll
    for (int g = 0; g < 8; ++g) {
      const int query = q0 + g + 8 * h;
      float v0 = sc0[g], v1 = sc1[g];
      if (key0 > query) v0 = -__builtin_inff();
      if (key1 > query) v1 = -__builtin_inff();

      float mx = fmaxf(v0, v1);
      mx = fmaxf(mx, __shfl_xor(mx, 1, 32));
      mx = fmaxf(mx, __shfl_xor(mx, 2, 32));
      mx = fmaxf(mx, __shfl_xor(mx, 4, 32));
      mx = fmaxf(mx, __shfl_xor(mx, 8, 32));

      const float mnew = fmaxf(mrow[g], mx);
      const float p0 = __expf(v0 - mnew);
      const float p1 = __expf(v1 - mnew);
      float rs = p0 + p1;
      rs += __shfl_xor(rs, 1, 32);
      rs += __shfl_xor(rs, 2, 32);
      rs += __shfl_xor(rs, 4, 32);
      rs += __shfl_xor(rs, 8, 32);

      const float scl = __expf(mrow[g] - mnew);
      lrow[g] = lrow[g] * scl + rs;
      mrow[g] = mnew;
#pragma unroll
      for (int nt = 0; nt < 8; ++nt) o[nt][g] *= scl;

      const int row = g + 8 * h;
      plds[wave][row * 32 + r]      = (__bf16)p0;
      plds[wave][row * 32 + 16 + r] = (__bf16)p1;
    }

    asm volatile("s_wait_dscnt 0" ::: "memory");
    v16bf pf = load_a_frag(&plds[wave][r * 32], h);

#pragma unroll
    for (int nt = 0; nt < 8; ++nt) o[nt] = wmma_bf16(pf, vf[nt], o[nt]);
    asm volatile("" ::: "memory");
  }
#endif

  const int b = bh >> 4, head = bh & 15;
#pragma unroll
  for (int nt = 0; nt < 8; ++nt) {
#pragma unroll
    for (int g = 0; g < 8; ++g) {
      const int s = q0 + g + 8 * h;
      const int e = head * HEAD_DIM + nt * 16 + r;
      og[((size_t)b * SEQ + s) * N_EMBD + e] = (__bf16)(o[nt][g] / lrow[g]);
    }
  }
}

// ---------------------------------------------------------------------------
// Output projection: [4096,2048] @ [2048,2048] + bias -> fp32 d_out
// ---------------------------------------------------------------------------
__global__ __launch_bounds__(256) void proj_gemm(
    const __bf16* __restrict__ A, const __bf16* __restrict__ Wt,
    const float* __restrict__ bias, float* __restrict__ out) {
  const int nbN = N_EMBD / 128;
  const int m0 = (blockIdx.x / nbN) * 128;
  const int n0 = (blockIdx.x % nbN) * 128;

  v8f acc[2][4] = {};
  gemm_core_128x128(A + (size_t)m0 * N_EMBD, Wt + (size_t)n0 * N_EMBD,
                    N_EMBD, acc);

  const int lane = threadIdx.x & 31;
  const int r = lane & 15, h = lane >> 4;
  const int wave = threadIdx.x >> 5;
  const int wm = (wave & 3) * 32, wn = (wave >> 2) * 64;

#pragma unroll
  for (int mi = 0; mi < 2; ++mi) {
#pragma unroll
    for (int nt = 0; nt < 4; ++nt) {
      const int n = n0 + wn + nt * 16 + r;
      const float bv = bias[n];
#pragma unroll
      for (int g = 0; g < 8; ++g) {
        const int m = m0 + wm + mi * 16 + g + 8 * h;
        out[(size_t)m * N_EMBD + n] = acc[mi][nt][g] + bv;
      }
    }
  }
}

// ---------------------------------------------------------------------------
// Workspace layout (bytes)
// ---------------------------------------------------------------------------
#define OFF_XBF    ((size_t)0)
#define OFF_WQKVT  (OFF_XBF   + (size_t)MTOT * N_EMBD * 2)
#define OFF_WPROJT (OFF_WQKVT + (size_t)N_EMBD * N_QKV * 2)
#define OFF_Q      (OFF_WPROJT+ (size_t)N_EMBD * N_EMBD * 2)
#define OFF_K      (OFF_Q     + (size_t)MTOT * N_EMBD * 2)
#define OFF_VT     (OFF_K     + (size_t)MTOT * N_EMBD * 2)
#define OFF_AO     (OFF_VT    + (size_t)MTOT * N_EMBD * 2)

extern "C" void kernel_launch(void* const* d_in, const int* in_sizes, int n_in,
                              void* d_out, int out_size, void* d_ws, size_t ws_size,
                              hipStream_t stream) {
  const float* hs    = (const float*)d_in[0];
  const float* wqkv  = (const float*)d_in[1];
  const float* bqkv  = (const float*)d_in[2];
  const float* wproj = (const float*)d_in[3];
  const float* bproj = (const float*)d_in[4];
  float* out = (float*)d_out;

  char* ws = (char*)d_ws;
  __bf16* x_bf    = (__bf16*)(ws + OFF_XBF);
  __bf16* wqkv_t  = (__bf16*)(ws + OFF_WQKVT);
  __bf16* wproj_t = (__bf16*)(ws + OFF_WPROJT);
  __bf16* q_ws    = (__bf16*)(ws + OFF_Q);
  __bf16* k_ws    = (__bf16*)(ws + OFF_K);
  __bf16* vt_ws   = (__bf16*)(ws + OFF_VT);
  __bf16* ao_ws   = (__bf16*)(ws + OFF_AO);

  cvt_f32_bf16<<<2048, 256, 0, stream>>>(hs, x_bf, MTOT * N_EMBD);
  transpose_cvt<<<(N_EMBD / 32) * (N_QKV / 32), 256, 0, stream>>>(
      wqkv, wqkv_t, N_EMBD, N_QKV);
  transpose_cvt<<<(N_EMBD / 32) * (N_EMBD / 32), 256, 0, stream>>>(
      wproj, wproj_t, N_EMBD, N_EMBD);

  qkv_gemm<<<(MTOT / 128) * (N_QKV / 128), 256, 0, stream>>>(
      x_bf, wqkv_t, bqkv, q_ws, k_ws, vt_ws);

  attn_kernel<<<BATCH * N_HEAD * (SEQ / 16) / 8, 256, 0, stream>>>(
      q_ws, k_ws, vt_ws, ao_ws);

  proj_gemm<<<(MTOT / 128) * (N_EMBD / 128), 256, 0, stream>>>(
      ao_ws, wproj_t, bproj, out);
}
